// PointImageFusion_68745246540465
// MI455X (gfx1250) — matrix operand order repaired and checked by reading, not compiled, not run on hardware
//
#include <hip/hip_runtime.h>

typedef __attribute__((ext_vector_type(16))) _Float16 v16h;
typedef __attribute__((ext_vector_type(8)))  _Float16 v8h;
typedef __attribute__((ext_vector_type(8)))  float    v8f;

#define VC 32
#define PD 8
#define CCH 64
#define FID 16
#define VFD 64
#define IMG_H 256
#define IMG_W 704

// ---------------- shared memory layout ----------------
struct WaveScratch {
    _Float16 A1[16 * 64];    // sampled image features tile (GEMM1 A), f16
    _Float16 Hh[16 * 32];    // GEMM1 output / GEMM2 A, K padded 16->32
    _Float16 Fu[16 * 96];    // fusion tile (GEMM3 A), K padded 81->96
    _Float16 G4[16 * 64];    // GEMM3 output / GEMM4 A
    float    validv[16];
    float    pad[8];         // keep 32B alignment of successive entries
};

struct SharedMem {
    _Float16 wt1[16 * 64];   // w_ip1^T  [n][k], k=0..63
    _Float16 wt2[16 * 32];   // w_ip2^T  [n][k], k padded to 32
    _Float16 wt3[64 * 96];   // w_fp1^T  [n][k], k padded 81->96
    _Float16 wt4[64 * 64];   // w_fp2^T  [n][k]
    WaveScratch wv[4];
};

// ---------------- fragment helpers (CDNA5 wave32 layouts) ----------------
// A-matrix 16x32 f16: lane L holds row M=L&15; element e holds
// K = kt*32 + 16*(L>>4) + e  -> 16 contiguous halves.
__device__ __forceinline__ v16h load_frag_A(const _Float16* base, int rowLen,
                                            int kt, int lane) {
    int m = lane & 15, h = lane >> 4;
    const _Float16* p = base + m * rowLen + kt * 32 + 16 * h;
    v8h lo = *(const v8h*)p;          // K = kt*32 + 16h .. +7
    v8h hi = *(const v8h*)(p + 8);    // K = kt*32 + 16h + 8 .. +7
    v16h r;
#pragma unroll
    for (int i = 0; i < 8; ++i) { r[i] = lo[i]; r[i + 8] = hi[i]; }
    return r;
}

// B-matrix 32x16 f16 from K-major transposed weights wt[N][Ktot]:
// lane L holds column N = nt*16 + (L&15); element e holds K = kt*32 + 16*(L>>4) + e.
__device__ __forceinline__ v16h load_frag_B(const _Float16* wt, int Ktot,
                                            int kt, int nt, int lane) {
    int n = nt * 16 + (lane & 15), h = lane >> 4;
    const _Float16* p = wt + n * Ktot + kt * 32 + 16 * h;
    return *(const v16h*)p;           // 16 contiguous halves, 32B aligned
}

__device__ __forceinline__ v8f wmma16(v16h a, v16h b, v8f c) {
    return __builtin_amdgcn_wmma_f32_16x16x32_f16(false, a, false, b,
                                                  (short)0, c, false, false);
}

// ---------------- fused kernel ----------------
__global__ void __launch_bounds__(128)
point_image_fusion_kernel(const float* __restrict__ voxel_features,
                          const float* __restrict__ voxels,
                          const float* __restrict__ img_feats,
                          const float* __restrict__ t_cam,
                          const float* __restrict__ proj,
                          const float* __restrict__ w_ip1, const float* __restrict__ b_ip1,
                          const float* __restrict__ w_ip2, const float* __restrict__ b_ip2,
                          const float* __restrict__ w_fp1, const float* __restrict__ b_fp1,
                          const float* __restrict__ w_fp2, const float* __restrict__ b_fp2,
                          const int* __restrict__ num_points,
                          const int* __restrict__ coors,
                          float* __restrict__ out) {
    __shared__ SharedMem s;
    const int tid = threadIdx.x;
    const int wid = tid >> 5;
    const int lane = tid & 31;

    // ---- stage all weights transposed/padded as f16 into LDS (once/block) ----
    for (int i = tid; i < 16 * 64; i += 128) {
        int n = i >> 6, k = i & 63;
        s.wt1[i] = (_Float16)w_ip1[k * FID + n];
    }
    for (int i = tid; i < 16 * 32; i += 128) {
        int n = i >> 5, k = i & 31;
        s.wt2[i] = (_Float16)(k < 16 ? w_ip2[k * FID + n] : 0.f);
    }
    for (int i = tid; i < 64 * 96; i += 128) {
        int n = i / 96, k = i - n * 96;
        s.wt3[i] = (_Float16)(k < 81 ? w_fp1[k * VFD + n] : 0.f);
    }
    for (int i = tid; i < 64 * 64; i += 128) {
        int n = i >> 6, k = i & 63;
        s.wt4[i] = (_Float16)w_fp2[k * VFD + n];
    }

    WaveScratch& Wv = s.wv[wid];
    const int vt = blockIdx.x * 64 + wid * 16;   // 16-voxel tile base for this wave

    // ---- phase 1: per-voxel reduce + project + bilinear gather (lane == point) ----
    for (int i = 0; i < 16; ++i) {
        const int v = vt + i;
        const float4* vp = (const float4*)(voxels + ((size_t)v * VC + lane) * PD);
        float4 f0 = vp[0];                 // x, y, z, misc0
        float4 f1 = vp[1];                 // misc1..3, flag
        int np = num_points[v];
        float msk = (lane < np && f1.w > 0.5f) ? 1.f : 0.f;
        float sx = f0.x * msk, sy = f0.y * msk, sz = f0.z * msk, sc = msk;
#pragma unroll
        for (int d = 16; d >= 1; d >>= 1) {
            sx += __shfl_xor(sx, d, 32);
            sy += __shfl_xor(sy, d, 32);
            sz += __shfl_xor(sz, d, 32);
            sc += __shfl_xor(sc, d, 32);
        }
        float vm = sc > 0.f ? 1.f : 0.f;
        float denom = fmaxf(sc, 1.f);
        float cx = sx / denom, cy = sy / denom, cz = sz / denom;

        int b = coors[v * 4];
        const float* T = t_cam + b * 12;
        float cam0 = T[0] * cx + T[1] * cy + T[2]  * cz + T[3];
        float cam1 = T[4] * cx + T[5] * cy + T[6]  * cz + T[7];
        float cam2 = T[8] * cx + T[9] * cy + T[10] * cz + T[11];
        const float* P = proj + b * 9;
        float uw = P[0] * cam0 + P[1] * cam1 + P[2] * cam2;
        float vw = P[3] * cam0 + P[4] * cam1 + P[5] * cam2;
        float ww = P[6] * cam0 + P[7] * cam1 + P[8] * cam2;
        float wz = fmaxf(ww, 1e-5f);
        float u  = (uw / wz) * 0.5f;   // W/ORI_W = 704/1408
        float vv = (vw / wz) * 0.5f;   // H/ORI_H = 256/512
        bool valid = (cam2 > 0.f) && (u >= 0.f) && (u <= (float)(IMG_W - 1)) &&
                     (vv >= 0.f) && (vv <= (float)(IMG_H - 1)) && (vm > 0.f);
        float validf = valid ? 1.f : 0.f;
        float uc = fminf(fmaxf(u, 0.f), (float)(IMG_W - 1));
        float vc = fminf(fmaxf(vv, 0.f), (float)(IMG_H - 1));
        float x0 = floorf(uc), y0 = floorf(vc);
        int x0i = (int)x0, y0i = (int)y0;
        int x1i = min(x0i + 1, IMG_W - 1), y1i = min(y0i + 1, IMG_H - 1);
        float wx = uc - x0, wy = vc - y0;

        const float* img = img_feats + (size_t)b * CCH * IMG_H * IMG_W;
        size_t p00 = (size_t)y0i * IMG_W + x0i, p01 = (size_t)y0i * IMG_W + x1i;
        size_t p10 = (size_t)y1i * IMG_W + x0i, p11 = (size_t)y1i * IMG_W + x1i;
#pragma unroll
        for (int cc = 0; cc < 2; ++cc) {
            int c = lane + cc * 32;
            const float* chan = img + (size_t)c * IMG_H * IMG_W;
            float f00 = chan[p00], f01 = chan[p01], f10 = chan[p10], f11 = chan[p11];
            float sval = ((f00 * (1.f - wx) + f01 * wx) * (1.f - wy) +
                          (f10 * (1.f - wx) + f11 * wx) * wy) * validf;
            Wv.A1[i * 64 + c] = (_Float16)sval;
            Wv.Fu[i * 96 + c] = (_Float16)voxel_features[(size_t)v * VFD + c];
        }
        if (lane == 0) {
            Wv.validv[i] = validf;
            Wv.Fu[i * 96 + 64] = (_Float16)validf;   // fusion valid column
        }
        if (lane < 15) Wv.Fu[i * 96 + 81 + lane] = (_Float16)0.f;  // K pad 81->96
    }
    // zero the K=16..31 pad of the GEMM2 A tile
    for (int j = lane; j < 16 * 16; j += 32) {
        int r = j >> 4, c = j & 15;
        Wv.Hh[r * 32 + 16 + c] = (_Float16)0.f;
    }
    __syncthreads();

    const int n0 = lane & 15, hh = lane >> 4;

    // ---- GEMM1: sampled[16x64] @ w_ip1[64x16] -> relu -> Hh ----
    {
        v8f acc = {};
        acc = wmma16(load_frag_A(Wv.A1, 64, 0, lane), load_frag_B(s.wt1, 64, 0, 0, lane), acc);
        acc = wmma16(load_frag_A(Wv.A1, 64, 1, lane), load_frag_B(s.wt1, 64, 1, 0, lane), acc);
        float bb = b_ip1[n0];
#pragma unroll
        for (int r = 0; r < 8; ++r) {
            int m = r + 8 * hh;
            Wv.Hh[m * 32 + n0] = (_Float16)fmaxf(acc[r] + bb, 0.f);
        }
    }
    __syncthreads();

    // ---- GEMM2: Hh[16x16] @ w_ip2[16x16] -> *validf -> fusion cols 65..80 ----
    {
        v8f acc = {};
        acc = wmma16(load_frag_A(Wv.Hh, 32, 0, lane), load_frag_B(s.wt2, 32, 0, 0, lane), acc);
        float bb = b_ip2[n0];
#pragma unroll
        for (int r = 0; r < 8; ++r) {
            int m = r + 8 * hh;
            float t = (acc[r] + bb) * Wv.validv[m];
            Wv.Fu[m * 96 + 65 + n0] = (_Float16)t;
        }
    }
    __syncthreads();

    // ---- GEMM3: fusion[16x96(81)] @ w_fp1[81x64] -> relu -> G4 ----
    {
        v8f acc[4] = {{}, {}, {}, {}};
#pragma unroll
        for (int kt = 0; kt < 3; ++kt) {
            v16h a = load_frag_A(Wv.Fu, 96, kt, lane);
#pragma unroll
            for (int nt = 0; nt < 4; ++nt)
                acc[nt] = wmma16(a, load_frag_B(s.wt3, 96, kt, nt, lane), acc[nt]);
        }
#pragma unroll
        for (int nt = 0; nt < 4; ++nt) {
            float bb = b_fp1[nt * 16 + n0];
#pragma unroll
            for (int r = 0; r < 8; ++r) {
                int m = r + 8 * hh;
                Wv.G4[m * 64 + nt * 16 + n0] = (_Float16)fmaxf(acc[nt][r] + bb, 0.f);
            }
        }
    }
    __syncthreads();

    // ---- GEMM4: G4[16x64] @ w_fp2[64x64] + bias + residual -> out ----
    {
        v8f acc[4] = {{}, {}, {}, {}};
#pragma unroll
        for (int kt = 0; kt < 2; ++kt) {
            v16h a = load_frag_A(Wv.G4, 64, kt, lane);
#pragma unroll
            for (int nt = 0; nt < 4; ++nt)
                acc[nt] = wmma16(a, load_frag_B(s.wt4, 64, kt, nt, lane), acc[nt]);
        }
#pragma unroll
        for (int nt = 0; nt < 4; ++nt) {
            float bb = b_fp2[nt * 16 + n0];
#pragma unroll
            for (int r = 0; r < 8; ++r) {
                int m = r + 8 * hh;
                size_t off = (size_t)(vt + m) * VFD + nt * 16 + n0;
                out[off] = voxel_features[off] + acc[nt][r] + bb;
            }
        }
    }
}

extern "C" void kernel_launch(void* const* d_in, const int* in_sizes, int n_in,
                              void* d_out, int out_size, void* d_ws, size_t ws_size,
                              hipStream_t stream) {
    const float* voxel_features = (const float*)d_in[0];
    const float* voxels         = (const float*)d_in[1];
    const float* img_feats      = (const float*)d_in[2];
    const float* t_cam          = (const float*)d_in[3];
    const float* proj           = (const float*)d_in[4];
    const float* w_ip1          = (const float*)d_in[5];
    const float* b_ip1          = (const float*)d_in[6];
    const float* w_ip2          = (const float*)d_in[7];
    const float* b_ip2          = (const float*)d_in[8];
    const float* w_fp1          = (const float*)d_in[9];
    const float* b_fp1          = (const float*)d_in[10];
    const float* w_fp2          = (const float*)d_in[11];
    const float* b_fp2          = (const float*)d_in[12];
    const int*   num_points     = (const int*)d_in[13];
    const int*   coors          = (const int*)d_in[14];
    float*       out            = (float*)d_out;

    const int N = in_sizes[0] / VFD;     // 40000
    const int blocks = N / 64;           // 625 (exact: 40000 = 64 * 625)

    point_image_fusion_kernel<<<blocks, 128, 0, stream>>>(
        voxel_features, voxels, img_feats, t_cam, proj,
        w_ip1, b_ip1, w_ip2, b_ip2, w_fp1, b_fp1, w_fp2, b_fp2,
        num_points, coors, out);
}
